// AutoEncoder_48790828483076
// MI455X (gfx1250) — matrix operand order repaired and checked by reading, not compiled, not run on hardware
//
#include <hip/hip_runtime.h>
#include <hip/hip_bf16.h>

#define BSETS 8192
#define NELEM 65536
#define DIMX  64
#define HIDX  64
#define MAXN  17
#define DMOD  128
#define NLAY  3
#define PHI_MID 72
#define SP_MID  40

typedef __attribute__((ext_vector_type(16))) _Float16 v16h;
typedef __attribute__((ext_vector_type(8)))  _Float16 v8h;
typedef __attribute__((ext_vector_type(8)))  float    v8f;

// ---------------- WMMA fragment loaders (CDNA5 16-bit layouts) ----------------
// A 16x32 (MxK): lanes 0-15 -> M=lane, lanes 16-31 -> M=lane-16.
// Elements 0..7 : K = 8*lh + 0..7 ; elements 8..15 : K = 16 + 8*lh + 0..7.
__device__ __forceinline__ v16h frag_a(const _Float16* base, int ldk, int lane) {
  int lm = lane & 15, lh = (lane >> 4) & 1;
  const _Float16* p = base + lm * ldk;
  v8h lo = *(const v8h*)(p + lh * 8);
  v8h hi = *(const v8h*)(p + 16 + lh * 8);
  return __builtin_shufflevector(lo, hi, 0,1,2,3,4,5,6,7,8,9,10,11,12,13,14,15);
}
// B 32x16 (KxN), staged transposed in LDS as wt[n][k] (k stride 1, row 32 halves).
// Elements 0..7 : K = 8*lh + 0..7 ; elements 8..15 : K = 16 + 8*lh + 0..7 (same as A).
__device__ __forceinline__ v16h frag_b(const _Float16* wt, int lane) {
  int lm = lane & 15, lh = (lane >> 4) & 1;
  const _Float16* p = wt + lm * 32;
  v8h lo = *(const v8h*)(p + lh * 8);
  v8h hi = *(const v8h*)(p + 16 + lh * 8);
  return __builtin_shufflevector(lo, hi, 0,1,2,3,4,5,6,7,8,9,10,11,12,13,14,15);
}

// ---------------- 64-thread (2-wave) workgroup GEMM: C[32xN] = A[32xK] * Wt^T + bias
// A: LDS f16 (row-major, ld=K). Wt: global f16, PRE-TRANSPOSED [N][K] (k contiguous).
// wch: LDS staging, N*32 halves (layout [n][k-chunk]). Writes f16 (out16, optional relu)
// or f32 (out32), ld = N. Chunk staging is pure b128 global->LDS copies.
template<int K, int N>
__device__ __forceinline__ void wg_gemm(const _Float16* __restrict__ A,
                                        const _Float16* __restrict__ Wt,
                                        const float* __restrict__ bias,
                                        _Float16* __restrict__ wch,
                                        _Float16* out16, float* out32,
                                        bool relu, int tid) {
  constexpr int NT = N / 16;
  const int wv = tid >> 5, lane = tid & 31;
  const int m0 = wv * 16;
  const int lm = lane & 15, lh = (lane >> 4) & 1;
  v8f acc[NT] = {};
  for (int kc = 0; kc < K; kc += 32) {
    __syncthreads();
#pragma unroll
    for (int i = tid; i < N * 4; i += 64) {       // 32xN chunk in v8h units
      int n = i >> 2, jj = i & 3;
      *(v8h*)(wch + n * 32 + jj * 8) =
          *(const v8h*)(Wt + (size_t)n * K + kc + jj * 8);
    }
    __syncthreads();
    v16h a = frag_a(A + m0 * K + kc, K, lane);
#pragma unroll
    for (int nt = 0; nt < NT; ++nt) {
      v16h b = frag_b(wch + nt * 16 * 32, lane);
      acc[nt] = __builtin_amdgcn_wmma_f32_16x16x32_f16(
          false, a, false, b, (short)0, acc[nt], false, false);
    }
  }
  __syncthreads();
#pragma unroll
  for (int nt = 0; nt < NT; ++nt) {
    int n = nt * 16 + lm;
    float bv = bias[n];
#pragma unroll
    for (int r = 0; r < 8; ++r) {
      int m = m0 + lh * 8 + r;                    // C/D: vgpr r -> M = r + 8*lh
      float v = acc[nt][r] + bv;
      if (relu) v = fmaxf(v, 0.0f);
      if (out16) out16[m * N + n] = (_Float16)v;
      else       out32[m * N + n] = v;
    }
  }
}

// ================= Kernel 0: weight prep — f32 [K][N] -> f16 transposed [N][K] ========
// Batched: BATCH matrices of KxN, each transposed independently.
__global__ void __launch_bounds__(256)
wtrans_kernel(const float* __restrict__ src, _Float16* __restrict__ dst,
              int batch, int K, int N) {
  int i = blockIdx.x * 256 + threadIdx.x;
  int total = batch * K * N;
  if (i >= total) return;
  int b = i / (K * N), r = i % (K * N);
  int k = r / N, n = r % N;
  dst[(size_t)b * K * N + (size_t)n * K + k] = (_Float16)src[i];
}

// ================= Kernel 1: psi MLP + segment sum (WMMA) =================
// 64 threads, 32 input rows (= 4 sets of 8) per WG.
__global__ void __launch_bounds__(64)
enc_kernel(const float* __restrict__ x, const int* __restrict__ batch,
           const _Float16* __restrict__ W1t, const float* __restrict__ b1,
           const float* __restrict__ g,  const float* __restrict__ beta,
           const _Float16* __restrict__ W2t, const float* __restrict__ b2,
           float* __restrict__ y2, int* __restrict__ ntrue) {
  __shared__ __align__(16) unsigned char smem[20480];
  _Float16* x16 = (_Float16*)(smem);            // 32x64
  _Float16* wch = (_Float16*)(smem + 4096);     // 64x32 transposed chunk
  float*    t32 = (float*)(smem + 8192);        // 32x64
  _Float16* t16 = (_Float16*)(smem + 16384);    // 32x64
  const int tid = threadIdx.x;
  const int r0 = blockIdx.x * 32;

  for (int i = tid; i < 32 * 64; i += 64)
    x16[i] = (_Float16)x[(size_t)r0 * 64 + i];

  wg_gemm<64, 64>(x16, W1t, b1, wch, nullptr, t32, false, tid);
  __syncthreads();
  if (tid < 32) {                                // LayerNorm + tanh, one row each
    float m = 0.f;
    for (int c = 0; c < 64; ++c) m += t32[tid * 64 + c];
    m *= (1.0f / 64.0f);
    float v = 0.f;
    for (int c = 0; c < 64; ++c) { float d = t32[tid * 64 + c] - m; v += d * d; }
    v *= (1.0f / 64.0f);
    float rs = rsqrtf(v + 1e-5f);
    for (int c = 0; c < 64; ++c)
      t16[tid * 64 + c] = (_Float16)tanhf((t32[tid * 64 + c] - m) * rs * g[c] + beta[c]);
  }
  wg_gemm<64, 64>(t16, W2t, b2, wch, nullptr, t32, false, tid);
  __syncthreads();
  // segment sum: 4 sets x 64 cols
  for (int i = tid; i < 256; i += 64) {
    int sl = i >> 6, c = i & 63;
    int gset = batch[r0 + 8 * sl];
    float s = 0.f;
    for (int j = 0; j < 8; ++j) s += t32[(8 * sl + j) * 64 + c];
    y2[(size_t)gset * 64 + c] = s;
  }
  if (tid < 4) {
    int gset = batch[r0 + 8 * tid];
    int cnt = 0;
    for (int j = 0; j < 8; ++j) cnt += (batch[r0 + 8 * tid + j] == gset) ? 1 : 0;
    ntrue[gset] = cnt;
  }
}

// ================= Kernel 2: phi + size-pred + h0 build (VALU; ~0.3% of FLOPs) =========
// 128 threads = 4 waves, one set per wave.
__global__ void __launch_bounds__(128)
phi_kernel(const float* __restrict__ y2, const int* __restrict__ ntrue,
           const float* __restrict__ pW1, const float* __restrict__ pb1,
           const float* __restrict__ pW2, const float* __restrict__ pb2,
           const float* __restrict__ sW1, const float* __restrict__ sb1,
           const float* __restrict__ sg,  const float* __restrict__ sbt,
           const float* __restrict__ sW2, const float* __restrict__ sb2,
           const float* __restrict__ mu,  const float* __restrict__ sigma,
           const float* __restrict__ noise,
           float* __restrict__ h, int* __restrict__ npred, float* __restrict__ maskout) {
  __shared__ float ybuf[4][81], pbuf[4][72], zbuf[4][64], sbuf[4][40];
  __shared__ float stat[4][2];
  __shared__ int   npsh[4];
  const int tid = threadIdx.x, wv = tid >> 5, lane = tid & 31;
  const int s = blockIdx.x * 4 + wv;

  for (int c = lane; c < 64; c += 32) ybuf[wv][c] = y2[(size_t)s * 64 + c];
  int nt = ntrue[s];
  for (int c = 64 + lane; c < 81; c += 32) ybuf[wv][c] = ((c - 64) == nt) ? 1.0f : 0.0f;
  __syncthreads();
  for (int c = lane; c < PHI_MID; c += 32) {     // phi layer 1 (tanh)
    float a = pb1[c];
    for (int k = 0; k < 81; ++k) a += ybuf[wv][k] * pW1[k * PHI_MID + c];
    pbuf[wv][c] = tanhf(a);
  }
  __syncthreads();
  for (int c = lane; c < 64; c += 32) {          // phi layer 2 -> z
    float a = pb2[c];
    for (int k = 0; k < PHI_MID; ++k) a += pbuf[wv][k] * pW2[k * 64 + c];
    zbuf[wv][c] = a;
  }
  __syncthreads();
  for (int c = lane; c < SP_MID; c += 32) {      // sp layer 1
    float a = sb1[c];
    for (int k = 0; k < 64; ++k) a += zbuf[wv][k] * sW1[k * SP_MID + c];
    sbuf[wv][c] = a;
  }
  __syncthreads();
  if (lane == 0) {                               // LN stats over 40
    float m = 0.f;
    for (int c = 0; c < SP_MID; ++c) m += sbuf[wv][c];
    m /= SP_MID;
    float v = 0.f;
    for (int c = 0; c < SP_MID; ++c) { float d = sbuf[wv][c] - m; v += d * d; }
    stat[wv][0] = m; stat[wv][1] = v / SP_MID;
  }
  __syncthreads();
  float mm = stat[wv][0], rs = rsqrtf(stat[wv][1] + 1e-5f);
  for (int c = lane; c < SP_MID; c += 32)
    sbuf[wv][c] = tanhf((sbuf[wv][c] - mm) * rs * sg[c] + sbt[c]);
  __syncthreads();
  if (lane == 0) {                               // logits + argmax (first max)
    float best = -1e30f; int bi = 0;
    for (int c = 0; c < MAXN; ++c) {
      float a = sb2[c];
      for (int k = 0; k < SP_MID; ++k) a += sbuf[wv][k] * sW2[k * MAXN + c];
      if (a > best) { best = a; bi = c; }
    }
    npsh[wv] = bi; npred[s] = bi;
  }
  __syncthreads();
  int np = npsh[wv];
  for (int i = lane; i < MAXN * DMOD; i += 32) { // h0 = [x0 | z] * fmask
    int t = i >> 7, c = i & 127;
    float fm = (t < np) ? 1.0f : 0.0f;
    float val = (c < 64)
        ? (mu[c] + fabsf(sigma[c]) * noise[((size_t)s * MAXN + t) * 64 + c])
        : zbuf[wv][c - 64];
    h[((size_t)s * MAXN + t) * DMOD + c] = val * fm;
  }
  for (int t = lane; t < MAXN; t += 32)
    maskout[(size_t)s * MAXN + t] = (t < np) ? 1.0f : 0.0f;
}

// ================= Kernel 3: fused 3-layer transformer + conv (WMMA) =================
// 64 threads, one set (17 tokens padded to 32 rows). Sets are layer-independent,
// so all 3 layers + output conv run in one kernel with no global sync.
__global__ void __launch_bounds__(64)
tf_kernel(const float* __restrict__ h, const int* __restrict__ npred,
          const _Float16* __restrict__ Wq, const float* __restrict__ bq,
          const _Float16* __restrict__ Wk, const float* __restrict__ bk,
          const _Float16* __restrict__ Wv, const float* __restrict__ bv,
          const _Float16* __restrict__ Wo, const float* __restrict__ bo,
          const float* __restrict__ ln1g, const float* __restrict__ ln1b,
          const float* __restrict__ ln2g, const float* __restrict__ ln2b,
          const _Float16* __restrict__ fW1, const float* __restrict__ fb1,
          const _Float16* __restrict__ fW2, const float* __restrict__ fb2,
          const _Float16* __restrict__ cW,  const float* __restrict__ cb,
          float* __restrict__ xr) {
  __shared__ __align__(16) unsigned char smem[57344];
  float*    h32 = (float*)(smem);               // 32x128 f32 residual stream
  _Float16* h16 = (_Float16*)(smem + 16384);    // 32x128 f16 (A input)
  _Float16* wch = (_Float16*)(smem + 24576);    // 128x32 transposed weight chunk
  _Float16* q16 = (_Float16*)(smem + 32768);    // Q, then attn-out, then FFN mid
  _Float16* k16 = (_Float16*)(smem + 40960);
  _Float16* v16 = (_Float16*)(smem + 49152);
  float*  tmp32 = (float*)(smem + 40960);       // aliases K|V (dead when used)

  const int tid = threadIdx.x;
  const int set = blockIdx.x;
  const float* hsrc = h + (size_t)set * MAXN * DMOD;
  const int np = npred[set];

  for (int i = tid; i < 32 * DMOD; i += 64) {
    int t = i >> 7;
    float v = (t < MAXN) ? hsrc[i] : 0.0f;
    h32[i] = v; h16[i] = (_Float16)v;
  }

  for (int L = 0; L < NLAY; ++L) {
    const size_t o2 = (size_t)L * DMOD * DMOD, o1 = (size_t)L * DMOD;
    // QKV projections
    wg_gemm<128, 128>(h16, Wq + o2, bq + o1, wch, q16, nullptr, false, tid);
    wg_gemm<128, 128>(h16, Wk + o2, bk + o1, wch, k16, nullptr, false, tid);
    wg_gemm<128, 128>(h16, Wv + o2, bv + o1, wch, v16, nullptr, false, tid);
    __syncthreads();
    // attention: 2 heads x 17 query rows (masked softmax), o overwrites q16
    if (tid < 2 * MAXN) {
      int hd = tid / MAXN, qt = tid % MAXN;
      float att[MAXN], mx = -1e30f;
      for (int k = 0; k < MAXN; ++k) {
        float sc = 0.f;
        for (int d = 0; d < 64; ++d)
          sc += (float)q16[qt * DMOD + hd * 64 + d] * (float)k16[k * DMOD + hd * 64 + d];
        sc *= 0.125f;                            // 1/sqrt(64)
        if (k >= np) sc -= 1e9f;
        att[k] = sc; mx = fmaxf(mx, sc);
      }
      float den = 0.f;
      for (int k = 0; k < MAXN; ++k) { att[k] = __expf(att[k] - mx); den += att[k]; }
      float inv = 1.0f / den;
      for (int d = 0; d < 64; ++d) {
        float o = 0.f;
        for (int k = 0; k < MAXN; ++k) o += att[k] * (float)v16[k * DMOD + hd * 64 + d];
        q16[qt * DMOD + hd * 64 + d] = (_Float16)(o * inv);
      }
    }
    // O projection (tmp32 aliases dead K/V), then residual + LN1
    wg_gemm<128, 128>(q16, Wo + o2, bo + o1, wch, nullptr, tmp32, false, tid);
    __syncthreads();
    if (tid < 32) {
      int t = tid;
      if (t < MAXN) {
        float m = 0.f;
        for (int c = 0; c < DMOD; ++c) m += h32[t * DMOD + c] + tmp32[t * DMOD + c];
        m *= (1.0f / DMOD);
        float v = 0.f;
        for (int c = 0; c < DMOD; ++c) {
          float d = h32[t * DMOD + c] + tmp32[t * DMOD + c] - m; v += d * d;
        }
        float rs = rsqrtf(v * (1.0f / DMOD) + 1e-5f);
        for (int c = 0; c < DMOD; ++c) {
          float val = (h32[t * DMOD + c] + tmp32[t * DMOD + c] - m) * rs * ln1g[o1 + c] + ln1b[o1 + c];
          h32[t * DMOD + c] = val; h16[t * DMOD + c] = (_Float16)val;
        }
      } else {
        for (int c = 0; c < DMOD; ++c) { h32[t * DMOD + c] = 0.f; h16[t * DMOD + c] = (_Float16)0.f; }
      }
    }
    // FFN: relu(h@fW1+b1)@fW2+b2, residual + LN2
    wg_gemm<128, 64>(h16, fW1 + (size_t)L * DMOD * 64, fb1 + (size_t)L * 64,
                     wch, q16, nullptr, true, tid);
    wg_gemm<64, 128>(q16, fW2 + (size_t)L * 64 * DMOD, fb2 + o1,
                     wch, nullptr, tmp32, false, tid);
    __syncthreads();
    if (tid < 32) {
      int t = tid;
      if (t < MAXN) {
        float m = 0.f;
        for (int c = 0; c < DMOD; ++c) m += h32[t * DMOD + c] + tmp32[t * DMOD + c];
        m *= (1.0f / DMOD);
        float v = 0.f;
        for (int c = 0; c < DMOD; ++c) {
          float d = h32[t * DMOD + c] + tmp32[t * DMOD + c] - m; v += d * d;
        }
        float rs = rsqrtf(v * (1.0f / DMOD) + 1e-5f);
        for (int c = 0; c < DMOD; ++c) {
          float val = (h32[t * DMOD + c] + tmp32[t * DMOD + c] - m) * rs * ln2g[o1 + c] + ln2b[o1 + c];
          h32[t * DMOD + c] = val; h16[t * DMOD + c] = (_Float16)val;
        }
      } else {
        for (int c = 0; c < DMOD; ++c) { h32[t * DMOD + c] = 0.f; h16[t * DMOD + c] = (_Float16)0.f; }
      }
    }
  }
  // output 1x1 conv + mask
  wg_gemm<128, 64>(h16, cW, cb, wch, nullptr, tmp32, false, tid);
  __syncthreads();
  for (int i = tid; i < MAXN * 64; i += 64) {
    int t = i >> 6, c = i & 63;
    float fm = (t < np) ? 1.0f : 0.0f;
    xr[((size_t)set * MAXN + t) * 64 + c] = tmp32[t * 64 + c] * fm;
  }
}

// ============================ launcher ============================
extern "C" void kernel_launch(void* const* d_in, const int* in_sizes, int n_in,
                              void* d_out, int out_size, void* d_ws, size_t ws_size,
                              hipStream_t stream) {
  const float* x      = (const float*)d_in[0];
  const int*   batch  = (const int*)d_in[1];
  const float* psiW1  = (const float*)d_in[2];
  const float* psib1  = (const float*)d_in[3];
  const float* psig   = (const float*)d_in[4];
  const float* psibt  = (const float*)d_in[5];
  const float* psiW2  = (const float*)d_in[6];
  const float* psib2  = (const float*)d_in[7];
  const float* phiW1  = (const float*)d_in[8];
  const float* phib1  = (const float*)d_in[9];
  const float* phiW2  = (const float*)d_in[10];
  const float* phib2  = (const float*)d_in[11];
  const float* spW1   = (const float*)d_in[12];
  const float* spb1   = (const float*)d_in[13];
  const float* spg    = (const float*)d_in[14];
  const float* spbt   = (const float*)d_in[15];
  const float* spW2   = (const float*)d_in[16];
  const float* spb2   = (const float*)d_in[17];
  const float* mu     = (const float*)d_in[18];
  const float* sigma  = (const float*)d_in[19];
  const float* noise  = (const float*)d_in[20];
  const float* tWq    = (const float*)d_in[21];
  const float* tbq    = (const float*)d_in[22];
  const float* tWk    = (const float*)d_in[23];
  const float* tbk    = (const float*)d_in[24];
  const float* tWv    = (const float*)d_in[25];
  const float* tbv    = (const float*)d_in[26];
  const float* tWo    = (const float*)d_in[27];
  const float* tbo    = (const float*)d_in[28];
  const float* ln1g   = (const float*)d_in[29];
  const float* ln1b   = (const float*)d_in[30];
  const float* ln2g   = (const float*)d_in[31];
  const float* ln2b   = (const float*)d_in[32];
  const float* fW1    = (const float*)d_in[33];
  const float* fb1    = (const float*)d_in[34];
  const float* fW2    = (const float*)d_in[35];
  const float* fb2    = (const float*)d_in[36];
  const float* convW  = (const float*)d_in[37];
  const float* convb  = (const float*)d_in[38];

  // ---- workspace layout (16B-aligned sections) ----
  char* ws = (char*)d_ws;
  float* y2    = (float*)ws;                                   // B*64 f32   (2 MB)
  int*   ntrue = (int*)(ws + (size_t)BSETS * 64 * 4);          // B
  int*   nprd  = ntrue + BSETS;                                // B
  float* hbuf  = (float*)(nprd + BSETS);                       // B*17*128 f32
  _Float16* f16base = (_Float16*)((char*)hbuf + (size_t)BSETS * MAXN * DMOD * 4);
  _Float16* psiW1t = f16base;                                  // 64*64
  _Float16* psiW2t = psiW1t + 64 * 64;                         // 64*64
  _Float16* Wqt    = psiW2t + 64 * 64;                         // 3*128*128
  _Float16* Wkt    = Wqt + NLAY * DMOD * DMOD;
  _Float16* Wvt    = Wkt + NLAY * DMOD * DMOD;
  _Float16* Wot    = Wvt + NLAY * DMOD * DMOD;
  _Float16* fW1t   = Wot + NLAY * DMOD * DMOD;                 // 3*(128x64 -> 64x128)
  _Float16* fW2t   = fW1t + NLAY * DMOD * 64;                  // 3*(64x128 -> 128x64)
  _Float16* convWt = fW2t + NLAY * 64 * DMOD;                  // 128x64 -> 64x128

  float* xr      = (float*)d_out;                              // B*17*64
  float* maskout = xr + (size_t)BSETS * MAXN * DIMX;           // B*17

  // ---- weight prep: f32 [K][N] -> f16 transposed [N][K] ----
  auto tr = [&](const float* s, _Float16* d, int b, int K, int N) {
    int total = b * K * N;
    wtrans_kernel<<<(total + 255) / 256, 256, 0, stream>>>(s, d, b, K, N);
  };
  tr(psiW1, psiW1t, 1, 64, 64);
  tr(psiW2, psiW2t, 1, 64, 64);
  tr(tWq, Wqt, NLAY, DMOD, DMOD);
  tr(tWk, Wkt, NLAY, DMOD, DMOD);
  tr(tWv, Wvt, NLAY, DMOD, DMOD);
  tr(tWo, Wot, NLAY, DMOD, DMOD);
  tr(fW1, fW1t, NLAY, DMOD, 64);
  tr(fW2, fW2t, NLAY, 64, DMOD);
  tr(convW, convWt, 1, DMOD, 64);

  enc_kernel<<<NELEM / 32, 64, 0, stream>>>(x, batch, psiW1t, psib1, psig, psibt,
                                            psiW2t, psib2, y2, ntrue);
  phi_kernel<<<BSETS / 4, 128, 0, stream>>>(y2, ntrue, phiW1, phib1, phiW2, phib2,
                                            spW1, spb1, spg, spbt, spW2, spb2,
                                            mu, sigma, noise, hbuf, nprd, maskout);
  tf_kernel<<<BSETS, 64, 0, stream>>>(hbuf, nprd, Wqt, tbq, Wkt, tbk, Wvt, tbv,
                                      Wot, tbo, ln1g, ln1b, ln2g, ln2b,
                                      fW1t, fb1, fW2t, fb2, convWt, convb, xr);
  (void)in_sizes; (void)n_in; (void)out_size; (void)ws_size;
}
